// MaxViTStyleBevBlock_2259152797915
// MI455X (gfx1250) — matrix operand order repaired and checked by reading, not compiled, not run on hardware
//
#include <hip/hip_runtime.h>

typedef __attribute__((ext_vector_type(16))) _Float16 v16h;
typedef __attribute__((ext_vector_type(8)))  _Float16 v8h;
typedef __attribute__((ext_vector_type(8)))  float    v8f;

#define DEVINL __device__ __forceinline__

constexpr int NB = 2, ND = 128, NH = 224, NWD = 224;
constexpr int NHID = 512, NSEH = 128;
constexpr int NHEADS = 8, NDH = 64, NINNER = 512, NWIN = 7;
constexpr int NFFH = 512;
constexpr int HWSZ = NH * NWD;        // 50176
constexpr int NPIX = NB * HWSZ;       // 100352
constexpr float EPSV = 1e-5f;

DEVINL float geluf(float x) { return 0.5f * x * (1.0f + erff(x * 0.70710678118654752f)); }

// ---- WMMA fragment helpers (16x16x32 f16, wave32; contiguous k order used by both A and B) ----
// A (16x32, MxK): lane m = lane&15 ; elems 0..15 -> K = (lane>>4)*16 + 0..15
//                 (two adjacent 16B chunks)
DEVINL v16h load_A(const _Float16* A, int lda, int lane) {
  const _Float16* r = A + (size_t)(lane & 15) * lda + ((lane >> 4) << 4);
  v8h lo = *(const v8h*)(r);
  v8h hi = *(const v8h*)(r + 8);
  return __builtin_shufflevector(lo, hi, 0,1,2,3,4,5,6,7,8,9,10,11,12,13,14,15);
}
// B (32x16, KxN) supplied as B^T rows (N-major, K contiguous): lane n = lane&15,
// K base = (lane>>4)*16 -> one contiguous 32B load per lane.
DEVINL v16h load_BT(const _Float16* BT, int ldb, int lane) {
  const _Float16* r = BT + (size_t)(lane & 15) * ldb + ((lane >> 4) << 4);
  return *(const v16h*)(r);
}
DEVINL v8f wmma_k(const _Float16* A, int lda, const _Float16* BT, int ldb,
                  int K, int lane, v8f c) {
  for (int k0 = 0; k0 < K; k0 += 32) {
    v16h a = load_A(A + k0, lda, lane);
    v16h b = load_BT(BT + k0, ldb, lane);
    c = __builtin_amdgcn_wmma_f32_16x16x32_f16(false, a, false, b, (short)0, c, false, false);
  }
  return c;
}

// ---------------- small utility kernels ----------------
__global__ void k_cvt(const float* __restrict__ s, _Float16* __restrict__ d, int n) {
  for (int i = blockIdx.x * blockDim.x + threadIdx.x; i < n; i += gridDim.x * blockDim.x)
    d[i] = (_Float16)s[i];
}
__global__ void k_zero(float* __restrict__ p, int n) {
  for (int i = blockIdx.x * blockDim.x + threadIdx.x; i < n; i += gridDim.x * blockDim.x)
    p[i] = 0.f;
}

// ---------------- MBConv: conv1 (1x1, 128->512) + BN + GELU ----------------
__global__ __launch_bounds__(128) void k_conv1(
    const float* __restrict__ x, const _Float16* __restrict__ w1f,
    const float* __restrict__ g, const float* __restrict__ bb,
    const float* __restrict__ mm, const float* __restrict__ vv,
    _Float16* __restrict__ h1) {
  __shared__ __align__(64) _Float16 As[64 * 128];
  const int tid = threadIdx.x;
  const int pix0 = blockIdx.x * 64;
  const int bIdx = pix0 / HWSZ;
  const int hw0 = pix0 - bIdx * HWSZ;
  for (int e = tid; e < 64 * 128; e += 128) {
    int row = e & 63, c = e >> 6;
    As[row * 128 + c] = (_Float16)x[(size_t)(bIdx * ND + c) * HWSZ + hw0 + row];
  }
  __syncthreads();
  const int wave = tid >> 5, lane = tid & 31;
  for (int t = wave; t < 4 * 32; t += 4) {
    int m0 = (t & 3) * 16, n0 = (t >> 2) * 16;
    v8f c8 = {};
    c8 = wmma_k(As + m0 * 128, 128, w1f + (size_t)n0 * 128, 128, 128, lane, c8);
    int n = n0 + (lane & 15);
    int mb = m0 + ((lane >> 4) << 3);
    float inv  = g[n] * rsqrtf(vv[n] + EPSV);
    float beta = bb[n] - mm[n] * inv;
    _Float16* dst = h1 + (size_t)(bIdx * NHID + n) * HWSZ + hw0;
#pragma unroll
    for (int r = 0; r < 8; ++r)
      dst[mb + r] = (_Float16)geluf(c8[r] * inv + beta);
  }
}

// ---------------- MBConv: depthwise 3x3 + BN + GELU + SE partial sums ----------------
__global__ __launch_bounds__(256) void k_dw(
    const _Float16* __restrict__ h1, const float* __restrict__ dw,
    const float* __restrict__ g, const float* __restrict__ bb,
    const float* __restrict__ mm, const float* __restrict__ vv,
    _Float16* __restrict__ h2, float* __restrict__ pool) {
  __shared__ float red[256];
  const int plane = blockIdx.x / 196;          // b*512 + c
  const int hw = (blockIdx.x % 196) * 256 + threadIdx.x;
  const int c = plane % NHID;
  const int y = hw / NWD, xx = hw % NWD;
  const _Float16* src = h1 + (size_t)plane * HWSZ;
  float acc = 0.f;
#pragma unroll
  for (int dy = 0; dy < 3; ++dy)
#pragma unroll
    for (int dx = 0; dx < 3; ++dx) {
      int yy = y + dy - 1, xs = xx + dx - 1;
      if (yy >= 0 && yy < NH && xs >= 0 && xs < NWD)
        acc += (float)src[yy * NWD + xs] * dw[c * 9 + dy * 3 + dx];
    }
  float inv = g[c] * rsqrtf(vv[c] + EPSV);
  float val = geluf(acc * inv + (bb[c] - mm[c] * inv));
  h2[(size_t)plane * HWSZ + hw] = (_Float16)val;
  red[threadIdx.x] = val;
  __syncthreads();
  for (int s = 128; s > 0; s >>= 1) {
    if (threadIdx.x < s) red[threadIdx.x] += red[threadIdx.x + s];
    __syncthreads();
  }
  if (threadIdx.x == 0) atomicAdd(&pool[plane], red[0]);
}

// ---------------- SE gate: sigmoid(relu(mean @ se1^T) @ se2^T) ----------------
__global__ __launch_bounds__(128) void k_se(
    const float* __restrict__ pool, const float* __restrict__ se1,
    const float* __restrict__ se2, float* __restrict__ sg) {
  __shared__ float hid[NB * NSEH];
  const int tid = threadIdx.x;  // 128
  const float invHW = 1.f / (float)HWSZ;
  for (int b = 0; b < NB; ++b) {
    float a = 0.f;
    for (int k = 0; k < NHID; ++k)
      a += (pool[b * NHID + k] * invHW) * se1[tid * NHID + k];
    hid[b * NSEH + tid] = fmaxf(a, 0.f);
  }
  __syncthreads();
  for (int o = tid; o < NHID; o += 128)
    for (int b = 0; b < NB; ++b) {
      float a = 0.f;
      for (int j = 0; j < NSEH; ++j) a += hid[b * NSEH + j] * se2[o * NSEH + j];
      sg[b * NHID + o] = 1.f / (1.f + expf(-a));
    }
}

// ---------------- MBConv: conv3 (1x1, 512->128) + BN3, trunk = 2x + h3 ----------------
__global__ __launch_bounds__(128) void k_conv3(
    const float* __restrict__ x, const _Float16* __restrict__ h2,
    const float* __restrict__ sg, const _Float16* __restrict__ w3f,
    const float* __restrict__ g, const float* __restrict__ bb,
    const float* __restrict__ mm, const float* __restrict__ vv,
    float* __restrict__ trunk) {
  extern __shared__ __align__(64) char dyn3[];
  _Float16* As = (_Float16*)dyn3;               // 64 x 512 f16 = 64KB
  const int tid = threadIdx.x;
  const int pix0 = blockIdx.x * 64;
  const int bIdx = pix0 / HWSZ;
  const int hw0 = pix0 - bIdx * HWSZ;
  for (int e = tid; e < 64 * 512; e += 128) {
    int row = e & 63, c = e >> 6;
    float s = sg[bIdx * NHID + c];
    As[row * 512 + c] =
        (_Float16)((float)h2[(size_t)(bIdx * NHID + c) * HWSZ + hw0 + row] * s);
  }
  __syncthreads();
  const int wave = tid >> 5, lane = tid & 31;
  for (int t = wave; t < 4 * 8; t += 4) {
    int m0 = (t & 3) * 16, n0 = (t >> 2) * 16;
    v8f c8 = {};
    c8 = wmma_k(As + m0 * 512, 512, w3f + (size_t)n0 * 512, 512, 512, lane, c8);
    int n = n0 + (lane & 15);
    int mb = m0 + ((lane >> 4) << 3);
    float inv  = g[n] * rsqrtf(vv[n] + EPSV);
    float beta = bb[n] - mm[n] * inv;
    const float* xr = x + (size_t)(bIdx * ND + n) * HWSZ + hw0;
    float* dst = trunk + (size_t)(bIdx * ND + n) * HWSZ + hw0;
#pragma unroll
    for (int r = 0; r < 8; ++r)
      dst[mb + r] = 2.f * xr[mb + r] + (c8[r] * inv + beta);
  }
}

// ---------------- fused window attention + FFN ----------------
DEVINL void ln_rows(const float* accs, _Float16* xns,
                    const float* g, const float* b, int tid) {
  if (tid < 64) {
    int r = tid;
    float mu = 0.f;
    for (int c = 0; c < 128; ++c) mu += accs[r * 128 + c];
    mu *= (1.f / 128.f);
    float var = 0.f;
    for (int c = 0; c < 128; ++c) { float d = accs[r * 128 + c] - mu; var += d * d; }
    var *= (1.f / 128.f);
    float inv = rsqrtf(var + EPSV);
    if (r < 49)
      for (int c = 0; c < 128; ++c)
        xns[r * 128 + c] = (_Float16)(((accs[r * 128 + c] - mu) * inv) * g[c] + b[c]);
    else
      for (int c = 0; c < 128; ++c) xns[r * 128 + c] = (_Float16)0.f;
  }
}

__global__ __launch_bounds__(128) void k_attn(
    const float* tin, float* tout,
    const float* __restrict__ lng, const float* __restrict__ lnb,
    const _Float16* __restrict__ qkvf, const _Float16* __restrict__ owf,
    const float* __restrict__ ob, const float* __restrict__ rel,
    const float* __restrict__ lng2, const float* __restrict__ lnb2,
    const _Float16* __restrict__ w1f, const float* __restrict__ b1,
    const _Float16* __restrict__ w2f, const float* __restrict__ b2) {
  extern __shared__ __align__(64) char dyn[];
  float*    accs = (float*)dyn;                       // 64x128 f32 (32KB)
  _Float16* xns  = (_Float16*)(dyn + 32768);          // 64x128 f16 (16KB)
  _Float16* sbuf = (_Float16*)(dyn + 49152);          // 3 x 64x64 f16 (24KB)
  _Float16* qo = sbuf;                                // q, later o
  _Float16* kv = sbuf + 4096;                         // k, later v^T
  _Float16* pp = sbuf + 8192;                         // sim / softmax P
  _Float16* hb = sbuf;                                // FFN hidden chunk 64x128

  const int tid = threadIdx.x;
  const int wave = tid >> 5, lane = tid & 31;
  const int widx = blockIdx.x;
  const int bIdx = widx >> 10;
  const int rem = widx & 1023;
  const int wy = rem >> 5, wx = rem & 31;
  const int pbase = (wy * NWIN) * NWD + wx * NWIN;

  { // load token tile (49x128, zero-pad to 64)
    int c = tid;
    const float* src = tin + (size_t)(bIdx * ND + c) * HWSZ + pbase;
    for (int r = 0; r < 49; ++r)
      accs[r * 128 + c] = src[(r / 7) * NWD + (r % 7)];
    for (int r = 49; r < 64; ++r) accs[r * 128 + c] = 0.f;
  }
  __syncthreads();

  // ===== attention =====
  ln_rows(accs, xns, lng, lnb, tid);
  __syncthreads();

  for (int h = 0; h < NHEADS; ++h) {
    // q and k (xn @ Wq_h^T / Wk_h^T)
    for (int t = wave; t < 32; t += 4) {
      int which = t >> 4, tt = t & 15;
      int m0 = (tt & 3) * 16, n0 = (tt >> 2) * 16;
      const _Float16* wb = qkvf + (size_t)(which * NINNER + h * NDH + n0) * ND;
      v8f c8 = {};
      c8 = wmma_k(xns + m0 * 128, 128, wb, 128, 128, lane, c8);
      _Float16* dst = (which == 0) ? qo : kv;
      int n = lane & 15, mb = m0 + ((lane >> 4) << 3);
#pragma unroll
      for (int r = 0; r < 8; ++r) dst[(mb + r) * 64 + n0 + n] = (_Float16)c8[r];
    }
    __syncthreads();
    // sim = q @ k^T * scale + rel_bias
    for (int t = wave; t < 16; t += 4) {
      int m0 = (t & 3) * 16, n0 = (t >> 2) * 16;
      v8f c8 = {};
      c8 = wmma_k(qo + m0 * 64, 64, kv + n0 * 64, 64, 64, lane, c8);
      int j = n0 + (lane & 15), mb = m0 + ((lane >> 4) << 3);
#pragma unroll
      for (int r = 0; r < 8; ++r) {
        int i = mb + r;
        float val = -60000.f;
        if (i < 49 && j < 49) {
          int ridx = ((i / 7) - (j / 7) + 6) * 13 + ((i % 7) - (j % 7) + 6);
          val = c8[r] * 0.125f + rel[ridx * NHEADS + h];
        }
        pp[i * 64 + j] = (_Float16)val;
      }
    }
    __syncthreads();
    // softmax (one thread per row)
    if (tid < 64) {
      int i = tid;
      if (i < 49) {
        float mx = -1e30f;
        for (int j = 0; j < 49; ++j) mx = fmaxf(mx, (float)pp[i * 64 + j]);
        float s = 0.f;
        for (int j = 0; j < 49; ++j) s += expf((float)pp[i * 64 + j] - mx);
        float is = 1.f / s;
        for (int j = 0; j < 49; ++j)
          pp[i * 64 + j] = (_Float16)(expf((float)pp[i * 64 + j] - mx) * is);
        for (int j = 49; j < 64; ++j) pp[i * 64 + j] = (_Float16)0.f;
      } else {
        for (int j = 0; j < 64; ++j) pp[i * 64 + j] = (_Float16)0.f;
      }
    }
    __syncthreads();
    // v (stored transposed into kv: vT[d][j])
    for (int t = wave; t < 16; t += 4) {
      int m0 = (t & 3) * 16, n0 = (t >> 2) * 16;
      const _Float16* wb = qkvf + (size_t)(2 * NINNER + h * NDH + n0) * ND;
      v8f c8 = {};
      c8 = wmma_k(xns + m0 * 128, 128, wb, 128, 128, lane, c8);
      int n = lane & 15, mb = m0 + ((lane >> 4) << 3);
#pragma unroll
      for (int r = 0; r < 8; ++r) kv[(n0 + n) * 64 + mb + r] = (_Float16)c8[r];
    }
    __syncthreads();
    // o = P @ v   (into qo)
    for (int t = wave; t < 16; t += 4) {
      int m0 = (t & 3) * 16, n0 = (t >> 2) * 16;
      v8f c8 = {};
      c8 = wmma_k(pp + m0 * 64, 64, kv + n0 * 64, 64, 64, lane, c8);
      int n = lane & 15, mb = m0 + ((lane >> 4) << 3);
#pragma unroll
      for (int r = 0; r < 8; ++r) qo[(mb + r) * 64 + n0 + n] = (_Float16)c8[r];
    }
    __syncthreads();
    // acc += o @ ow_h^T (+ ob once)
    for (int t = wave; t < 32; t += 4) {
      int m0 = (t & 3) * 16, n0 = (t >> 2) * 16;
      v8f c8 = {};
      c8 = wmma_k(qo + m0 * 64, 64, owf + (size_t)n0 * NINNER + h * NDH, NINNER,
                  64, lane, c8);
      int n = n0 + (lane & 15), mb = m0 + ((lane >> 4) << 3);
      float bias = (h == 0) ? ob[n] : 0.f;
#pragma unroll
      for (int r = 0; r < 8; ++r) accs[(mb + r) * 128 + n] += c8[r] + bias;
    }
    __syncthreads();
  }

  // ===== FFN =====
  ln_rows(accs, xns, lng2, lnb2, tid);
  __syncthreads();
  for (int f0 = 0; f0 < NFFH; f0 += 128) {
    // hb = gelu(xn @ w1_chunk^T + b1)
    for (int t = wave; t < 32; t += 4) {
      int m0 = (t & 3) * 16, n0 = (t >> 2) * 16;
      v8f c8 = {};
      c8 = wmma_k(xns + m0 * 128, 128, w1f + (size_t)(f0 + n0) * ND, ND, 128, lane, c8);
      int n = n0 + (lane & 15), mb = m0 + ((lane >> 4) << 3);
      float bias = b1[f0 + n];
#pragma unroll
      for (int r = 0; r < 8; ++r)
        hb[(mb + r) * 128 + n] = (_Float16)geluf(c8[r] + bias);
    }
    __syncthreads();
    // acc += hb @ w2_chunk^T (+ b2 once)
    for (int t = wave; t < 32; t += 4) {
      int m0 = (t & 3) * 16, n0 = (t >> 2) * 16;
      v8f c8 = {};
      c8 = wmma_k(hb + m0 * 128, 128, w2f + (size_t)n0 * NFFH + f0, NFFH, 128, lane, c8);
      int n = n0 + (lane & 15), mb = m0 + ((lane >> 4) << 3);
      float bias = (f0 == 0) ? b2[n] : 0.f;
#pragma unroll
      for (int r = 0; r < 8; ++r) accs[(mb + r) * 128 + n] += c8[r] + bias;
    }
    __syncthreads();
  }

  { // writeback: out = x_orig + tokens_final
    int c = tid;
    const float* src = tin + (size_t)(bIdx * ND + c) * HWSZ + pbase;
    float* dst = tout + (size_t)(bIdx * ND + c) * HWSZ + pbase;
    for (int r = 0; r < 49; ++r) {
      int o = (r / 7) * NWD + (r % 7);
      dst[o] = src[o] + accs[r * 128 + c];
    }
  }
}

// ---------------- host launcher ----------------
extern "C" void kernel_launch(void* const* d_in, const int* in_sizes, int n_in,
                              void* d_out, int out_size, void* d_ws, size_t ws_size,
                              hipStream_t stream) {
  (void)in_sizes; (void)n_in; (void)out_size; (void)ws_size;
  const float* x       = (const float*)d_in[0];
  const float* mb_w1   = (const float*)d_in[1];
  const float* bn1g    = (const float*)d_in[2];
  const float* bn1b    = (const float*)d_in[3];
  const float* bn1m    = (const float*)d_in[4];
  const float* bn1v    = (const float*)d_in[5];
  const float* mb_dw   = (const float*)d_in[6];
  const float* bn2g    = (const float*)d_in[7];
  const float* bn2b    = (const float*)d_in[8];
  const float* bn2m    = (const float*)d_in[9];
  const float* bn2v    = (const float*)d_in[10];
  const float* se1     = (const float*)d_in[11];
  const float* se2     = (const float*)d_in[12];
  const float* mb_w3   = (const float*)d_in[13];
  const float* bn3g    = (const float*)d_in[14];
  const float* bn3b    = (const float*)d_in[15];
  const float* bn3m    = (const float*)d_in[16];
  const float* bn3v    = (const float*)d_in[17];
  const float* ba_ln_g = (const float*)d_in[18];
  const float* ba_ln_b = (const float*)d_in[19];
  const float* ba_qkv  = (const float*)d_in[20];
  const float* ba_ow   = (const float*)d_in[21];
  const float* ba_ob   = (const float*)d_in[22];
  const float* ba_rel  = (const float*)d_in[23];
  const float* bf_ln_g = (const float*)d_in[24];
  const float* bf_ln_b = (const float*)d_in[25];
  const float* bf_w1   = (const float*)d_in[26];
  const float* bf_b1   = (const float*)d_in[27];
  const float* bf_w2   = (const float*)d_in[28];
  const float* bf_b2   = (const float*)d_in[29];
  const float* ga_ln_g = (const float*)d_in[30];
  const float* ga_ln_b = (const float*)d_in[31];
  const float* ga_qkv  = (const float*)d_in[32];
  const float* ga_ow   = (const float*)d_in[33];
  const float* ga_ob   = (const float*)d_in[34];
  const float* ga_rel  = (const float*)d_in[35];
  const float* gf_ln_g = (const float*)d_in[36];
  const float* gf_ln_b = (const float*)d_in[37];
  const float* gf_w1   = (const float*)d_in[38];
  const float* gf_b1   = (const float*)d_in[39];
  const float* gf_w2   = (const float*)d_in[40];
  const float* gf_b2   = (const float*)d_in[41];
  float* out = (float*)d_out;

  char* ws = (char*)d_ws;
  size_t off = 0;
  auto take = [&](size_t bytes) -> char* {
    char* p = ws + off;
    off += (bytes + 255) & ~(size_t)255;
    return p;
  };
  _Float16* w1f    = (_Float16*)take((size_t)NHID * ND * 2);
  _Float16* w3f    = (_Float16*)take((size_t)ND * NHID * 2);
  _Float16* baqkvf = (_Float16*)take((size_t)3 * NINNER * ND * 2);
  _Float16* baowf  = (_Float16*)take((size_t)ND * NINNER * 2);
  _Float16* bfw1f  = (_Float16*)take((size_t)NFFH * ND * 2);
  _Float16* bfw2f  = (_Float16*)take((size_t)ND * NFFH * 2);
  _Float16* gaqkvf = (_Float16*)take((size_t)3 * NINNER * ND * 2);
  _Float16* gaowf  = (_Float16*)take((size_t)ND * NINNER * 2);
  _Float16* gfw1f  = (_Float16*)take((size_t)NFFH * ND * 2);
  _Float16* gfw2f  = (_Float16*)take((size_t)ND * NFFH * 2);
  _Float16* h1     = (_Float16*)take((size_t)NB * NHID * HWSZ * 2);
  _Float16* h2     = (_Float16*)take((size_t)NB * NHID * HWSZ * 2);
  float*    pool   = (float*)take((size_t)NB * NHID * 4);
  float*    sgate  = (float*)take((size_t)NB * NHID * 4);
  float*    trunkA = (float*)take((size_t)NB * ND * HWSZ * 4);

  // weight conversions f32 -> f16
  k_cvt<<<256, 256, 0, stream>>>(mb_w1, w1f, NHID * ND);
  k_cvt<<<256, 256, 0, stream>>>(mb_w3, w3f, ND * NHID);
  k_cvt<<<256, 256, 0, stream>>>(ba_qkv, baqkvf, 3 * NINNER * ND);
  k_cvt<<<256, 256, 0, stream>>>(ba_ow, baowf, ND * NINNER);
  k_cvt<<<256, 256, 0, stream>>>(bf_w1, bfw1f, NFFH * ND);
  k_cvt<<<256, 256, 0, stream>>>(bf_w2, bfw2f, ND * NFFH);
  k_cvt<<<256, 256, 0, stream>>>(ga_qkv, gaqkvf, 3 * NINNER * ND);
  k_cvt<<<256, 256, 0, stream>>>(ga_ow, gaowf, ND * NINNER);
  k_cvt<<<256, 256, 0, stream>>>(gf_w1, gfw1f, NFFH * ND);
  k_cvt<<<256, 256, 0, stream>>>(gf_w2, gfw2f, ND * NFFH);
  k_zero<<<4, 256, 0, stream>>>(pool, NB * NHID);

  // MBConv
  k_conv1<<<NPIX / 64, 128, 0, stream>>>(x, w1f, bn1g, bn1b, bn1m, bn1v, h1);
  k_dw<<<NB * NHID * 196, 256, 0, stream>>>(h1, mb_dw, bn2g, bn2b, bn2m, bn2v, h2, pool);
  k_se<<<1, 128, 0, stream>>>(pool, se1, se2, sgate);
  k_conv3<<<NPIX / 64, 128, 64 * 512 * 2, stream>>>(x, h2, sgate, w3f,
                                                    bn3g, bn3b, bn3m, bn3v, trunkA);

  // block attention + FFN, then grid attention + FFN (identical 7x7 partition)
  const size_t attn_lds = 32768 + 16384 + 24576;  // 72KB
  k_attn<<<NB * 32 * 32, 128, attn_lds, stream>>>(
      trunkA, out, ba_ln_g, ba_ln_b, baqkvf, baowf, ba_ob, ba_rel,
      bf_ln_g, bf_ln_b, bfw1f, bf_b1, bfw2f, bf_b2);
  k_attn<<<NB * 32 * 32, 128, attn_lds, stream>>>(
      out, out, ga_ln_g, ga_ln_b, gaqkvf, gaowf, ga_ob, ga_rel,
      gf_ln_g, gf_ln_b, gfw1f, gf_b1, gfw2f, gf_b2);
}
